// SSMLayer_35253091565779
// MI455X (gfx1250) — matrix operand order repaired and checked by reading, not compiled, not run on hardware
//
#include <hip/hip_runtime.h>

// ---------------------------------------------------------------------------
// S4 SSM layer, chunked state-space block processing for gfx1250 (wave32).
//   N = 64 (state), L = 16384, BATCH = 512, chunk C = 64, J = 256 chunks.
// Pipeline:
//   k1: discretize (Neumann inverse)           -> Ab, Bb, Cb (f32, ws)
//   k2: derive AbC=Ab^64, Binp, Cout, Toep(K)  -> bf16 matrices (ws)
//   k2b: u -> bf16 pre-pass                    -> Ubf (ws)
//   k3: per-batch chunk scan (WMMA bf16, TDM)  -> Xall states (bf16, ws)
//   k4: per-chunk output (WMMA bf16, TDM)      -> y = Toep*u + Cout*x + D*u
// ---------------------------------------------------------------------------

#define LSEQ  16384
#define NBAT  512
#define CH    64
#define JCH   256

typedef __attribute__((ext_vector_type(16))) __bf16        v16bf;
typedef __attribute__((ext_vector_type(16))) unsigned short v16us;
typedef __attribute__((ext_vector_type(8)))  float          v8f;
typedef __attribute__((ext_vector_type(4)))  float          v4f;
typedef __attribute__((ext_vector_type(4)))  unsigned int   v4u;
typedef __attribute__((ext_vector_type(8)))  int            v8i;
typedef __attribute__((ext_vector_type(4)))  int            v4i;

union BF16x16 { v16bf b; v16us u; v4u q[2]; };

static __device__ __forceinline__ unsigned short f2bf(float f) {
  unsigned u = __float_as_uint(f);
  unsigned r = u + 0x7FFFu + ((u >> 16) & 1u);   // round-to-nearest-even
  return (unsigned short)(r >> 16);
}

// A-fragment (16x32 bf16) from row-major [M][64] bf16 matrix.
// lane m=lane&15, h=lane>>4: elems 0..15 -> K=k0+16h+{0..15}
static __device__ __forceinline__ v16bf load_a_frag(const unsigned short* M,
                                                    int m0, int k0, int lane) {
  int m = m0 + (lane & 15);
  int h = (lane >> 4) & 1;
  const unsigned short* p = M + m * 64 + k0 + h * 16;
  BF16x16 r;
  r.q[0] = *(const v4u*)p;
  r.q[1] = *(const v4u*)(p + 8);
  return r.b;
}

// B-fragment (32x16): per lane 16 contiguous K values of one column (bf16 src).
static __device__ __forceinline__ v16bf bfrag_from_bf16(const unsigned short* p) {
  BF16x16 r;
  r.q[0] = *(const v4u*)p;
  r.q[1] = *(const v4u*)(p + 8);
  return r.b;
}

static __device__ __forceinline__ v8f wmma_bf16(v16bf a, v16bf b, v8f c) {
  return __builtin_amdgcn_wmma_f32_16x16x32_bf16(false, a, false, b,
                                                 (short)0, c, false, false);
}

// ---------------------------------------------------------------------------
// Tensor Data Mover: DMA a contiguous run of bf16 elements (1-row tile) from
// global memory into LDS. D# per CDNA5 ISA ch.8: group0 {count=1, lds_addr,
// global_addr, type=2}, group1 {data_size=1(2B), tensor_dim0=n, tensor_dim1=1,
// tile_dim0=n, tile_dim1=1, dim0_stride=n}. Groups 2/3 zero (<=2D tensor).
// ---------------------------------------------------------------------------
#if defined(__has_builtin)
#if __has_builtin(__builtin_amdgcn_tensor_load_to_lds) && \
    __has_builtin(__builtin_amdgcn_s_wait_tensorcnt)
#define HAVE_TDM 1
#endif
#endif
#ifndef HAVE_TDM
#define HAVE_TDM 0
#endif

#if HAVE_TDM
static __device__ __forceinline__ void tdm_load_bf16_row(
    const unsigned short* gsrc, unsigned lds_byte_off, unsigned nelem) {
  unsigned long long ga = (unsigned long long)gsrc;
  v4u g0;
  g0.x = 1u;                                        // count=1 (valid), user desc
  g0.y = lds_byte_off;                              // lds_addr (bytes)
  g0.z = (unsigned)(ga & 0xFFFFFFFFu);              // global_addr[31:0]
  g0.w = (unsigned)((ga >> 32) & 0x01FFFFFFu)       // global_addr[56:32]
         | 0x80000000u;                             // type=2 ("image")
  v8i g1;
  g1[0] = 0x00010000;                               // data_size=1 (2B), wg_mask=0
  g1[1] = (int)((nelem & 0xFFFFu) << 16);           // tensor_dim0[15:0] @ bits63:48
  g1[2] = (int)(((nelem >> 16) & 0xFFFFu)           // tensor_dim0[31:16]
                | (1u << 16));                      // tensor_dim1 = 1
  g1[3] = (int)((nelem & 0xFFFFu) << 16);           // tile_dim0 @ bits127:112
  g1[4] = 1;                                        // tile_dim1 = 1, tile_dim2 = 0
  g1[5] = (int)nelem;                               // tensor_dim0_stride[31:0]
  g1[6] = 0;                                        // stride0 hi, stride1 lo
  g1[7] = 0;
  v4i gz = {0, 0, 0, 0};
#if defined(__clang_major__) && (__clang_major__ >= 23)
  v8i gz8 = {0, 0, 0, 0, 0, 0, 0, 0};
  __builtin_amdgcn_tensor_load_to_lds(g0, g1, gz, gz, gz8, 0);
#else
  __builtin_amdgcn_tensor_load_to_lds(g0, g1, gz, gz, 0);
#endif
}
#endif

static __device__ __forceinline__ unsigned lds_offset_of(const void* p) {
  // Generic pointer to __shared__: low 32 bits are the in-LDS byte offset.
  return (unsigned)(unsigned long long)p;
}

// ---------------------------------------------------------------------------
// Kernel 1: bilinear discretization. BL = inv(I - hA) via Neumann-Horner
// (||hA|| ~ 0.1 so 32 terms hit fp32 eps). Ab = BL(I+hA), Bb = step*BL*B.
// ---------------------------------------------------------------------------
__global__ __launch_bounds__(256) void ssm_setup_kernel(
    const float* __restrict__ A, const float* __restrict__ Bvec,
    const float* __restrict__ Cvec, const float* __restrict__ log_step,
    float* __restrict__ wAb, float* __restrict__ wBb, float* __restrict__ wCb) {
  __shared__ float sM[4096], sP[4096], sT[4096];
  const int tid = threadIdx.x;
  const float step = expf(log_step[0]);
  const float h = 0.5f * step;
  for (int i = tid; i < 4096; i += 256) {
    sM[i] = h * A[i];
    sP[i] = ((i >> 6) == (i & 63)) ? 1.f : 0.f;
  }
  for (int it = 0; it < 32; ++it) {
    __syncthreads();
    for (int i = tid; i < 4096; i += 256) {
      int r = i >> 6, c = i & 63;
      float acc = 0.f;
      for (int k = 0; k < 64; ++k) acc += sM[r * 64 + k] * sP[k * 64 + c];
      sT[i] = acc;
    }
    __syncthreads();
    for (int i = tid; i < 4096; i += 256)
      sP[i] = (((i >> 6) == (i & 63)) ? 1.f : 0.f) + sT[i];
  }
  __syncthreads();
  for (int i = tid; i < 4096; i += 256) {   // Ab = P @ (I + M)
    int r = i >> 6, c = i & 63;
    float acc = sP[r * 64 + c];
    for (int k = 0; k < 64; ++k) acc += sP[r * 64 + k] * sM[k * 64 + c];
    wAb[i] = acc;
  }
  if (tid < 64) {
    float acc = 0.f;
    for (int k = 0; k < 64; ++k) acc += sP[tid * 64 + k] * Bvec[k];
    wBb[tid] = step * acc;
    wCb[tid] = Cvec[tid];
  }
}

// ---------------------------------------------------------------------------
// Kernel 2: derive bf16 operator matrices: AbC = Ab^64 (6 squarings),
// Binp[:,i] = Ab^{63-i} Bb, Cout[r,:] = Cb Ab^{r+1}, Toep[t][tau] = K[t-tau].
// ---------------------------------------------------------------------------
__global__ __launch_bounds__(256) void ssm_derive_kernel(
    const float* __restrict__ wAb, const float* __restrict__ wBb,
    const float* __restrict__ wCb,
    unsigned short* __restrict__ AbC_bf, unsigned short* __restrict__ Binp_bf,
    unsigned short* __restrict__ Cout_bf, unsigned short* __restrict__ Toep_bf) {
  __shared__ float sA[4096], sS[4096], sT[4096];
  __shared__ float sv[64], svn[64], sw[64], swn[64], sK[64], sBb[64];
  const int tid = threadIdx.x;
  for (int i = tid; i < 4096; i += 256) sA[i] = wAb[i];
  if (tid < 64) {
    sv[tid] = wBb[tid];
    sBb[tid] = wBb[tid];
    sw[tid] = wCb[tid];
  }
  __syncthreads();
  for (int i2 = 0; i2 < 64; ++i2) {          // Binp column (63-i) = Ab^i Bb
    if (tid < 64) {
      Binp_bf[tid * 64 + (63 - i2)] = f2bf(sv[tid]);
      float acc = 0.f;
      for (int k = 0; k < 64; ++k) acc += sA[tid * 64 + k] * sv[k];
      svn[tid] = acc;
    }
    __syncthreads();
    if (tid < 64) sv[tid] = svn[tid];
    __syncthreads();
  }
  for (int l = 0; l < 64; ++l) {             // K[l] = w_l.Bb ; Cout[l] = w_{l+1}
    if (tid == 0) {
      float acc = 0.f;
      for (int k = 0; k < 64; ++k) acc += sw[k] * sBb[k];
      sK[l] = acc;
    }
    if (tid < 64) {
      float acc = 0.f;
      for (int k = 0; k < 64; ++k) acc += sw[k] * sA[k * 64 + tid];
      swn[tid] = acc;
    }
    __syncthreads();
    if (tid < 64) {
      sw[tid] = swn[tid];
      Cout_bf[l * 64 + tid] = f2bf(swn[tid]);
    }
    __syncthreads();
  }
  for (int i = tid; i < 4096; i += 256) {    // lower-triangular Toeplitz
    int r = i >> 6, c = i & 63;
    Toep_bf[i] = (r >= c) ? f2bf(sK[r - c]) : (unsigned short)0;
  }
  for (int i = tid; i < 4096; i += 256) sS[i] = sA[i];
  __syncthreads();
  for (int sq = 0; sq < 6; ++sq) {           // AbC = Ab^64
    for (int i = tid; i < 4096; i += 256) {
      int r = i >> 6, c = i & 63;
      float acc = 0.f;
      for (int k = 0; k < 64; ++k) acc += sS[r * 64 + k] * sS[k * 64 + c];
      sT[i] = acc;
    }
    __syncthreads();
    for (int i = tid; i < 4096; i += 256) sS[i] = sT[i];
    __syncthreads();
  }
  for (int i = tid; i < 4096; i += 256) AbC_bf[i] = f2bf(sS[i]);
}

// ---------------------------------------------------------------------------
// Kernel 2b: u (f32) -> Ubf (bf16) pre-pass; 8 elements/thread, b128 traffic.
// ---------------------------------------------------------------------------
__global__ __launch_bounds__(256) void u_to_bf16_kernel(
    const float* __restrict__ u, unsigned short* __restrict__ Ubf) {
  size_t i = ((size_t)blockIdx.x * 256 + threadIdx.x) * 8;
  v4f a = *(const v4f*)(u + i);
  v4f b = *(const v4f*)(u + i + 4);
  v4u pk;
  pk.x = (unsigned)f2bf(a[0]) | ((unsigned)f2bf(a[1]) << 16);
  pk.y = (unsigned)f2bf(a[2]) | ((unsigned)f2bf(a[3]) << 16);
  pk.z = (unsigned)f2bf(b[0]) | ((unsigned)f2bf(b[1]) << 16);
  pk.w = (unsigned)f2bf(b[2]) | ((unsigned)f2bf(b[3]) << 16);
  *(v4u*)(Ubf + i) = pk;
}

// ---------------------------------------------------------------------------
// Kernel 3: chunk scan. One wave per 16 batches; 256 sequential chunk steps:
//   X_{j+1} = AbC @ X_j + Binp @ u_chunk   (64x64 @ 64x16, bf16 WMMA, f32 acc)
// AbC/Binp staged into LDS via TDM. X round-trips LDS as packed bf16
// (stride-72 rows: 16B-aligned, conflict-free). States stored to Xall.
// ---------------------------------------------------------------------------
__global__ __launch_bounds__(32) void ssm_scan_kernel(
    const unsigned short* __restrict__ Ubf,
    const unsigned short* __restrict__ AbC_bf,
    const unsigned short* __restrict__ Binp_bf,
    unsigned short* __restrict__ Xall_bf) {
  const int lane = threadIdx.x & 31;
  const int n = lane & 15;
  const int hf = lane >> 4;
  const int b0 = blockIdx.x * 16;

  __shared__ unsigned short sOps[8192];      // AbC @ 0, Binp @ 4096 (shorts)
  __shared__ unsigned short sXb[16 * 72];    // packed bf16 state staging

#if HAVE_TDM
  {
    unsigned base = lds_offset_of(sOps);
    tdm_load_bf16_row(AbC_bf, base, 4096);
    tdm_load_bf16_row(Binp_bf, base + 8192, 4096);
    __builtin_amdgcn_s_wait_tensorcnt((short)0);
  }
  const unsigned short* pAbC = sOps;
  const unsigned short* pBinp = sOps + 4096;
#else
  const unsigned short* pAbC = AbC_bf;
  const unsigned short* pBinp = Binp_bf;
#endif

  v16bf fA[4][2], fB[4][2];
#pragma unroll
  for (int mt = 0; mt < 4; ++mt)
#pragma unroll
    for (int kf = 0; kf < 2; ++kf) {
      fA[mt][kf] = load_a_frag(pAbC, mt * 16, kf * 32, lane);
      fB[mt][kf] = load_a_frag(pBinp, mt * 16, kf * 32, lane);
    }

  v8f X[4];
#pragma unroll
  for (int mt = 0; mt < 4; ++mt) {
    v8f z = {0.f, 0.f, 0.f, 0.f, 0.f, 0.f, 0.f, 0.f};
    X[mt] = z;
  }

  const unsigned short* ub = Ubf + (size_t)(b0 + n) * LSEQ;

  for (int j = 0; j < JCH; ++j) {
    // 1) publish current state as packed bf16: LDS (layout shuffle) + Xall
    unsigned short* xo =
        Xall_bf + ((size_t)j * NBAT + (size_t)(b0 + n)) * 64 + hf * 8;
#pragma unroll
    for (int mt = 0; mt < 4; ++mt) {
      v4u pk;
      pk.x = (unsigned)f2bf(X[mt][0]) | ((unsigned)f2bf(X[mt][1]) << 16);
      pk.y = (unsigned)f2bf(X[mt][2]) | ((unsigned)f2bf(X[mt][3]) << 16);
      pk.z = (unsigned)f2bf(X[mt][4]) | ((unsigned)f2bf(X[mt][5]) << 16);
      pk.w = (unsigned)f2bf(X[mt][6]) | ((unsigned)f2bf(X[mt][7]) << 16);
      *(v4u*)&sXb[n * 72 + mt * 16 + hf * 8] = pk;
      *(v4u*)(xo + mt * 16) = pk;
    }
    __syncthreads();
    // 2) D-layout -> B-frag layout via LDS (raw ds_load_b128, no converts)
    v16bf xbf[2];
#pragma unroll
    for (int kf = 0; kf < 2; ++kf)
      xbf[kf] = bfrag_from_bf16(&sXb[n * 72 + kf * 32 + hf * 16]);
    __syncthreads();
    // 3) u chunk B-frags (pre-converted bf16)
    const unsigned short* up = ub + j * CH;
    v16bf ubf[2];
#pragma unroll
    for (int kf = 0; kf < 2; ++kf)
      ubf[kf] = bfrag_from_bf16(up + kf * 32 + hf * 16);
    if (j + 1 < JCH) __builtin_prefetch(up + CH, 0, 0);
    // 4) X <- AbC@X + Binp@U
#pragma unroll
    for (int mt = 0; mt < 4; ++mt) {
      v8f acc = {0.f, 0.f, 0.f, 0.f, 0.f, 0.f, 0.f, 0.f};
      acc = wmma_bf16(fB[mt][0], ubf[0], acc);
      acc = wmma_bf16(fB[mt][1], ubf[1], acc);
      acc = wmma_bf16(fA[mt][0], xbf[0], acc);
      acc = wmma_bf16(fA[mt][1], xbf[1], acc);
      X[mt] = acc;
    }
  }
}

// ---------------------------------------------------------------------------
// Kernel 4: outputs. One wave per (16-batch group, chunk):
//   Y = Toep @ u_chunk + Cout @ x_j  (WMMA)  ;  y += D * u  (f32 epilogue)
// Toep/Cout staged into LDS once per block via TDM (wave 0), shared by 4 waves.
// ---------------------------------------------------------------------------
__global__ __launch_bounds__(128) void ssm_out_kernel(
    const float* __restrict__ u, const unsigned short* __restrict__ Ubf,
    const float* __restrict__ Dp,
    const unsigned short* __restrict__ Toep_bf,
    const unsigned short* __restrict__ Cout_bf,
    const unsigned short* __restrict__ Xall_bf, float* __restrict__ out) {
  const int lane = threadIdx.x & 31;
  const int unit = blockIdx.x * 4 + (threadIdx.x >> 5);  // 0..8191
  const int j = unit >> 5;                               // 0..255
  const int bg = unit & 31;                              // 0..31
  const int b0 = bg * 16;
  const int n = lane & 15;
  const int hf = lane >> 4;
  const float D0 = Dp[0];

  __shared__ unsigned short sOps[8192];      // Toep @ 0, Cout @ 4096 (shorts)
#if HAVE_TDM
  if (threadIdx.x < 32) {
    unsigned base = lds_offset_of(sOps);
    tdm_load_bf16_row(Toep_bf, base, 4096);
    tdm_load_bf16_row(Cout_bf, base + 8192, 4096);
    __builtin_amdgcn_s_wait_tensorcnt((short)0);
  }
  __syncthreads();
  const unsigned short* pToep = sOps;
  const unsigned short* pCout = sOps + 4096;
#else
  const unsigned short* pToep = Toep_bf;
  const unsigned short* pCout = Cout_bf;
#endif

  v16bf fT[4][2], fC[4][2];
#pragma unroll
  for (int mt = 0; mt < 4; ++mt)
#pragma unroll
    for (int kf = 0; kf < 2; ++kf) {
      fT[mt][kf] = load_a_frag(pToep, mt * 16, kf * 32, lane);
      fC[mt][kf] = load_a_frag(pCout, mt * 16, kf * 32, lane);
    }

  const unsigned short* ubrow = Ubf + (size_t)(b0 + n) * LSEQ + j * CH;
  v16bf ubf[2];
#pragma unroll
  for (int kf = 0; kf < 2; ++kf)
    ubf[kf] = bfrag_from_bf16(ubrow + kf * 32 + hf * 16);

  const unsigned short* xrow =
      Xall_bf + ((size_t)j * NBAT + (size_t)(b0 + n)) * 64;
  v16bf xbf[2];
#pragma unroll
  for (int kf = 0; kf < 2; ++kf)
    xbf[kf] = bfrag_from_bf16(xrow + kf * 32 + hf * 16);

  const float* urow = u + (size_t)(b0 + n) * LSEQ + j * CH;
#pragma unroll
  for (int mt = 0; mt < 4; ++mt) {
    v8f acc = {0.f, 0.f, 0.f, 0.f, 0.f, 0.f, 0.f, 0.f};
    acc = wmma_bf16(fC[mt][0], xbf[0], acc);
    acc = wmma_bf16(fC[mt][1], xbf[1], acc);
    acc = wmma_bf16(fT[mt][0], ubf[0], acc);
    acc = wmma_bf16(fT[mt][1], ubf[1], acc);
    const int t0 = mt * 16 + hf * 8;
    const float* ue = urow + t0;
    float* orow = out + (size_t)(b0 + n) * LSEQ + j * CH + t0;
    v4f y0, y1;
#pragma unroll
    for (int i = 0; i < 4; ++i) {
      y0[i] = acc[i] + D0 * ue[i];
      y1[i] = acc[4 + i] + D0 * ue[4 + i];
    }
    *(v4f*)orow = y0;
    *(v4f*)(orow + 4) = y1;
  }
}

// ---------------------------------------------------------------------------
extern "C" void kernel_launch(void* const* d_in, const int* in_sizes, int n_in,
                              void* d_out, int out_size, void* d_ws,
                              size_t ws_size, hipStream_t stream) {
  (void)in_sizes; (void)n_in; (void)out_size; (void)ws_size;
  const float* u  = (const float*)d_in[0];
  const float* A  = (const float*)d_in[1];
  const float* B  = (const float*)d_in[2];
  const float* C  = (const float*)d_in[3];
  const float* D  = (const float*)d_in[4];
  const float* ls = (const float*)d_in[5];

  char* ws = (char*)d_ws;
  float* wAb = (float*)(ws + 0);           // 4096 f32
  float* wBb = (float*)(ws + 16384);       // 64 f32
  float* wCb = (float*)(ws + 16640);       // 64 f32
  unsigned short* AbC_bf  = (unsigned short*)(ws + 17408);          // 4096 bf16
  unsigned short* Binp_bf = (unsigned short*)(ws + 17408 + 8192);
  unsigned short* Cout_bf = (unsigned short*)(ws + 17408 + 16384);
  unsigned short* Toep_bf = (unsigned short*)(ws + 17408 + 24576);
  unsigned short* Xall_bf = (unsigned short*)(ws + 65536);          // 16 MB
  unsigned short* Ubf     = (unsigned short*)(ws + 16842752);       // 16 MB

  ssm_setup_kernel<<<1, 256, 0, stream>>>(A, B, C, ls, wAb, wBb, wCb);
  ssm_derive_kernel<<<1, 256, 0, stream>>>(wAb, wBb, wCb, AbC_bf, Binp_bf,
                                           Cout_bf, Toep_bf);
  u_to_bf16_kernel<<<4096, 256, 0, stream>>>(u, Ubf);
  ssm_scan_kernel<<<32, 32, 0, stream>>>(Ubf, AbC_bf, Binp_bf, Xall_bf);
  ssm_out_kernel<<<2048, 128, 0, stream>>>(u, Ubf, D, Toep_bf, Cout_bf,
                                           Xall_bf, (float*)d_out);
}
